// CRF_31318901522846
// MI455X (gfx1250) — matrix operand order repaired and checked by reading, not compiled, not run on hardware
//
#include <hip/hip_runtime.h>

#define DEVINL __device__ __forceinline__

constexpr int CB = 256;   // batch
constexpr int CS = 2048;  // sequence
constexpr int CT = 48;    // tags

typedef _Float16 v16h __attribute__((ext_vector_type(16)));
typedef float    v8f  __attribute__((ext_vector_type(8)));

union Frag {
  v16h v;
  _Float16 h[16];
  unsigned u[8];
};

constexpr float LOG2E = 1.4426950408889634f;
constexpr float LN2f  = 0.6931471805599453f;

DEVINL float fexp2(float x) { return __builtin_amdgcn_exp2f(x); }
DEVINL float flog2(float x) { return __builtin_amdgcn_logf(x); }   // v_log_f32 = log2
DEVINL float frcp(float x)  { return __builtin_amdgcn_rcpf(x); }

// Cross half-wave swap (lane L <-> lane L^16) as pure VALU: v_permlanex16_b32
// with the identity crossover select pattern. No LDS, no DScnt waits.
DEVINL int xswap_i(int x) {
  return __builtin_amdgcn_permlanex16(x, x, 0x76543210, (int)0xFEDCBA98u, false, false);
}
DEVINL float xswap_f(float x) {
  return __builtin_bit_cast(float, xswap_i(__builtin_bit_cast(int, x)));
}

DEVINL unsigned packh2(float a, float b) {
  unsigned short ua = __builtin_bit_cast(unsigned short, (_Float16)a);
  unsigned short ub = __builtin_bit_cast(unsigned short, (_Float16)b);
  return (unsigned)ua | ((unsigned)ub << 16);
}

// Load a 48-float row into this lane's 24 slots (j = 16*mt + v + 8*hi).
DEVINL void load_row48(const float* __restrict__ p, int hi, float (&dst)[3][8]) {
#pragma unroll
  for (int mt = 0; mt < 3; ++mt) {
    const float4 x = *(const float4*)(p + 16 * mt + 8 * hi);
    const float4 y = *(const float4*)(p + 16 * mt + 8 * hi + 4);
    dst[mt][0] = x.x; dst[mt][1] = x.y; dst[mt][2] = x.z; dst[mt][3] = x.w;
    dst[mt][4] = y.x; dst[mt][5] = y.y; dst[mt][6] = y.z; dst[mt][7] = y.w;
  }
}

// ---------------------------------------------------------------------------
// Kernel 1: gold-path score -> ws[b]
// ---------------------------------------------------------------------------
__global__ void __launch_bounds__(256) crf_score_kernel(
    const float* __restrict__ emis, const long long* __restrict__ tags,
    const unsigned char* __restrict__ mask, const float* __restrict__ trans,
    const float* __restrict__ startT, const float* __restrict__ endT,
    float* __restrict__ score) {
  const int b = blockIdx.x;
  const int tid = threadIdx.x;
  __shared__ float sred[256];
  __shared__ int   cred[256];

  const long long* tgb = tags + (size_t)b * CS;
  const unsigned char* mkb = mask + (size_t)b * CS;
  const float* emb = emis + (size_t)b * CS * CT;

  float acc = 0.f;
  int cnt = 0;
  for (int s = tid; s < CS; s += 256) {
    const int tg = (int)tgb[s];
    const int mk = mkb[s] ? 1 : 0;
    cnt += mk;
    const float em = emb[(size_t)s * CT + tg];
    if (s == 0) {
      acc += startT[tg] + em;
    } else {
      const int tp = (int)tgb[s - 1];
      const float tr = trans[tp * CT + tg];
      acc += mk ? (tr + em) : 0.f;
    }
  }
  sred[tid] = acc;
  cred[tid] = cnt;
  __syncthreads();
  for (int o = 128; o > 0; o >>= 1) {
    if (tid < o) { sred[tid] += sred[tid + o]; cred[tid] += cred[tid + o]; }
    __syncthreads();
  }
  if (tid == 0) {
    const int last = cred[0] - 1;
    const int lt = (int)tgb[last];
    score[b] = sred[0] + endT[lt];
  }
}

// ---------------------------------------------------------------------------
// Kernel 2: scaled forward algorithm with WMMA.
// State per batch column b (lane pair (b, b+16), j = 16*mt + v + 8*hi):
//   D[j]  : f32 WMMA output of last accepted step
//   E[j]  : exp(em_state[j] - g)   (em_state = last accepted emission row)
//   g     : max_j em_state[j],  base : scalar log-offset
// Invariant: alpha[j] = ln(D[j]) + ln(E[j]) + g + base.
// Per step: B_i = D_i * E_i * rcp(maxD)  (f16, <= 1), D' = expT^T x B (WMMA),
//           base' = base + g + ln(maxD) + Mt.  No per-element logs in loop.
// ---------------------------------------------------------------------------
__global__ void __launch_bounds__(32) crf_forward_kernel(
    const float* __restrict__ emis, const unsigned char* __restrict__ mask,
    const float* __restrict__ trans, const float* __restrict__ startT,
    const float* __restrict__ endT, const float* __restrict__ score,
    float* __restrict__ out) {
  const int lane = threadIdx.x & 31;
  const int hi = lane >> 4;
  const bool lo = (hi == 0);
  const int bb = blockIdx.x * 16 + (lane & 15);
  const float* emB = emis + (size_t)bb * CS * CT;
  const unsigned char* mkB = mask + (size_t)bb * CS;

  // ---- global max of transitions (one-time) ----
  float Mt = trans[lane];
  for (int idx = lane + 32; idx < CT * CT; idx += 32) Mt = fmaxf(Mt, trans[idx]);
#pragma unroll
  for (int o = 16; o > 0; o >>= 1) Mt = fmaxf(Mt, __shfl_xor(Mt, o, 32));

  // ---- loop-invariant A fragments: A[j][i] = exp(trans[i][j] - Mt), f16 ----
  Frag Af[3][2];
#pragma unroll
  for (int mt = 0; mt < 3; ++mt) {
    const int j = 16 * mt + (lane & 15);
#pragma unroll
    for (int kt = 0; kt < 2; ++kt) {
#pragma unroll
      for (int e = 0; e < 16; ++e) {
        const int K = e + 16 * hi;
        const int i = 32 * kt + K;
        const int ii = (i < CT) ? i : 0;
        const float tv = trans[ii * CT + j];
        const float ev = (i < CT) ? fexp2((tv - Mt) * LOG2E) : 0.f;
        Af[mt][kt].h[e] = (_Float16)ev;
      }
    }
  }

  // ---- init from start_trans + emissions[:,0] ----
  float D[3][8], E[3][8];
  float base, g;
  {
    float st[3][8], e0[3][8];
    load_row48(startT, hi, st);
    load_row48(emB, hi, e0);
    float ms = st[0][0];
#pragma unroll
    for (int mt = 0; mt < 3; ++mt)
#pragma unroll
      for (int v = 0; v < 8; ++v) ms = fmaxf(ms, st[mt][v]);
    ms = fmaxf(ms, xswap_f(ms));
    float g0 = e0[0][0];
#pragma unroll
    for (int mt = 0; mt < 3; ++mt)
#pragma unroll
      for (int v = 0; v < 8; ++v) g0 = fmaxf(g0, e0[mt][v]);
    g0 = fmaxf(g0, xswap_f(g0));
#pragma unroll
    for (int mt = 0; mt < 3; ++mt)
#pragma unroll
      for (int v = 0; v < 8; ++v) {
        D[mt][v] = fexp2((st[mt][v] - ms) * LOG2E);
        E[mt][v] = fexp2((e0[mt][v] - g0) * LOG2E);
      }
    base = ms;
    g = g0;
  }

  // software-pipelined emission rows (distance 2)
  float emA[3][8], emC[3][8];
  load_row48(emB + (size_t)1 * CT, hi, emA);
  load_row48(emB + (size_t)2 * CT, hi, emC);

  auto step = [&](int t, float (&em)[3][8]) {
    const unsigned char mku = mkB[t];
    const bool mk = (mku != 0);
    if (t + 16 < CS)
      __builtin_prefetch(emB + (size_t)(t + 16) * CT + 24 * hi, 0, 1);

    // --- off-chain: stats of this step's emission row (prefetched) ---
    float gn = em[0][0];
#pragma unroll
    for (int mt = 0; mt < 3; ++mt)
#pragma unroll
      for (int v = 0; v < 8; ++v) gn = fmaxf(gn, em[mt][v]);
    gn = fmaxf(gn, xswap_f(gn));

    // --- on-chain: normalize state into f16 B operand ---
    float md = D[0][0];
#pragma unroll
    for (int mt = 0; mt < 3; ++mt)
#pragma unroll
      for (int v = 0; v < 8; ++v) md = fmaxf(md, D[mt][v]);
    md = fmaxf(md, xswap_f(md));
    const float r = frcp(md);
    const float ell = -flog2(r) * LN2f;  // == ln(maxD), consistent with r

    unsigned p[3][4];
#pragma unroll
    for (int mt = 0; mt < 3; ++mt)
#pragma unroll
      for (int u = 0; u < 4; ++u) {
        const float b0 = D[mt][2 * u]     * E[mt][2 * u]     * r;
        const float b1 = D[mt][2 * u + 1] * E[mt][2 * u + 1] * r;
        p[mt][u] = packh2(b0, b1);
      }

    Frag B0, B1;
#pragma unroll
    for (int u = 0; u < 4; ++u) {
      const unsigned s0 = (unsigned)xswap_i((int)p[0][u]);
      const unsigned s1 = (unsigned)xswap_i((int)p[1][u]);
      const unsigned s2 = (unsigned)xswap_i((int)p[2][u]);
      B0.u[u]     = lo ? p[0][u] : s1;       // i 0..7   | 16..23
      B0.u[u + 4] = lo ? s0      : p[1][u];  // i 8..15  | 24..31
      B1.u[u]     = lo ? p[2][u] : 0u;       // i 32..39 | 48..55 (pad)
      B1.u[u + 4] = lo ? s2      : 0u;       // i 40..47 | 56..63 (pad)
    }

    // D'[j][b] = sum_i expT[i][j] * B[i][b]  (3 M-tiles, K=64 via 2 WMMAs)
    v8f acc[3];
#pragma unroll
    for (int mt = 0; mt < 3; ++mt) {
      v8f c = {};
      c = __builtin_amdgcn_wmma_f32_16x16x32_f16(false, Af[mt][0].v, false, B0.v,
                                                 (short)0, c, false, false);
      c = __builtin_amdgcn_wmma_f32_16x16x32_f16(false, Af[mt][1].v, false, B1.v,
                                                 (short)0, c, false, false);
      acc[mt] = c;
    }

    // --- masked state acceptance ---
    const float nbase = base + g + ell + Mt;  // uses OLD g (matches E used)
    base = mk ? nbase : base;
    g = mk ? gn : g;
#pragma unroll
    for (int mt = 0; mt < 3; ++mt)
#pragma unroll
      for (int v = 0; v < 8; ++v) {
        D[mt][v] = mk ? acc[mt][v] : D[mt][v];
        E[mt][v] = mk ? fexp2((em[mt][v] - gn) * LOG2E) : E[mt][v];
      }

    // refill this buffer for step t+2 (after use -> WAR only)
    if (t + 2 < CS) load_row48(emB + (size_t)(t + 2) * CT, hi, em);
  };

  for (int t = 1; t < CS; t += 2) {
    step(t, emA);
    if (t + 1 < CS) step(t + 1, emC);
  }

  // ---- norm = base + g + logsumexp_j(ln(D_j*E_j) + end_trans[j]) ----
  float et[3][8];
  load_row48(endT, hi, et);
  float val[3][8];
#pragma unroll
  for (int mt = 0; mt < 3; ++mt)
#pragma unroll
    for (int v = 0; v < 8; ++v)
      val[mt][v] = flog2(D[mt][v] * E[mt][v]) * LN2f + et[mt][v];
  float z = val[0][0];
#pragma unroll
  for (int mt = 0; mt < 3; ++mt)
#pragma unroll
    for (int v = 0; v < 8; ++v) z = fmaxf(z, val[mt][v]);
  z = fmaxf(z, xswap_f(z));
  float sm = 0.f;
#pragma unroll
  for (int mt = 0; mt < 3; ++mt)
#pragma unroll
    for (int v = 0; v < 8; ++v) sm += fexp2((val[mt][v] - z) * LOG2E);
  sm += xswap_f(sm);
  const float norm = base + g + z + flog2(sm) * LN2f;
  if (lo) out[bb] = norm - score[bb];
}

// ---------------------------------------------------------------------------
extern "C" void kernel_launch(void* const* d_in, const int* in_sizes, int n_in,
                              void* d_out, int out_size, void* d_ws, size_t ws_size,
                              hipStream_t stream) {
  const float*         emis   = (const float*)d_in[0];
  const long long*     tags   = (const long long*)d_in[1];      // int64
  const unsigned char* mask   = (const unsigned char*)d_in[2];  // bool
  const float*         trans  = (const float*)d_in[3];
  const float*         startT = (const float*)d_in[4];
  const float*         endT   = (const float*)d_in[5];
  float* out = (float*)d_out;
  float* ws  = (float*)d_ws;  // ws[0..B): gold-path scores

  crf_score_kernel<<<CB, 256, 0, stream>>>(emis, tags, mask, trans, startT, endT, ws);
  crf_forward_kernel<<<CB / 16, 32, 0, stream>>>(emis, mask, trans, startT, endT, ws, out);
}
